// SelfAttention_60361470378319
// MI455X (gfx1250) — matrix operand order repaired and checked by reading, not compiled, not run on hardware
//
#include <hip/hip_runtime.h>
#include <hip/hip_fp16.h>

typedef __attribute__((ext_vector_type(16))) _Float16 v16h;
typedef __attribute__((ext_vector_type(8)))  _Float16 v8h;
typedef __attribute__((ext_vector_type(8)))  float    v8f;

#define B_  4
#define T_  2048
#define D_  1024
#define H_  16
#define HD_ 64

#define KLD 72   // padded LDS row stride (halfs) for 64-wide tiles: bank-conflict free
#define BLD 40   // padded LDS row stride (halfs) for 32-wide GEMM B tiles

// ---------------- fragment loaders (wave32, 16x16x32 f16 WMMA) ----------------
// A fragment (16xK tile, K-step=32), source row-major with leading dim `ld`:
// lane r=l&15 holds row (row0+r); half hf=l>>4:
//   a[0..7]  = K[k0 + hf*8 .. +7]
//   a[8..15] = K[k0 + 16 + hf*8 .. +7]
__device__ __forceinline__ v16h load_frag_a(const _Float16* base, int ld, int row0, int k0) {
  const int lane = threadIdx.x & 31;
  const int r = lane & 15, hf = lane >> 4;
  const _Float16* p = base + (size_t)(row0 + r) * ld + k0;
  v8h lo = *(const v8h*)(p + hf * 8);
  v8h hi = *(const v8h*)(p + 16 + hf * 8);
  v16h f;
#pragma unroll
  for (int i = 0; i < 8; ++i) { f[i] = lo[i]; f[i + 8] = hi[i]; }
  return f;
}

// B fragment (Kx16 tile, K-step=32). We hold B^T row-major (col n of B == row n
// of the source, contiguous in K). lane r holds column (col0+r); half hf:
//   b[0..7]  = K[k0 + hf*8 .. +7]
//   b[8..15] = K[k0 + 16 + hf*8 .. +7]
__device__ __forceinline__ v16h load_frag_b(const _Float16* base, int ld, int col0, int k0) {
  const int lane = threadIdx.x & 31;
  const int r = lane & 15, hf = lane >> 4;
  const _Float16* p = base + (size_t)(col0 + r) * ld + k0;
  v8h lo = *(const v8h*)(p + hf * 8);
  v8h hi = *(const v8h*)(p + 16 + hf * 8);
  v16h f;
#pragma unroll
  for (int i = 0; i < 8; ++i) { f[i] = lo[i]; f[i + 8] = hi[i]; }
  return f;
}

__device__ __forceinline__ v8f wmma32(v16h a, v16h b, v8f c) {
  return __builtin_amdgcn_wmma_f32_16x16x32_f16(false, a, false, b, (short)0, c, false, false);
}

// ---- gfx1250 async global->LDS copy (bypasses VGPRs, tracked by ASYNCcnt) ----
__device__ __forceinline__ void async_ld_b128(unsigned lds_off, const _Float16* gsrc) {
  asm volatile("global_load_async_to_lds_b128 %0, %1, off"
               :: "v"(lds_off), "v"(gsrc) : "memory");
}
__device__ __forceinline__ unsigned lds_off_of(const void* p) {
  // generic LDS pointer = {SHARED_BASE, lds_addr[31:0]} -> low 32 bits are the offset
  return (unsigned)(unsigned long long)p;
}
__device__ __forceinline__ void wait_async0() {
#if __has_builtin(__builtin_amdgcn_s_wait_asynccnt)
  __builtin_amdgcn_s_wait_asynccnt(0);
#else
  asm volatile("s_wait_asynccnt 0x0" ::: "memory");
#endif
}

// ---------------- f32 -> f16 convert (8-wide) ----------------
__global__ void f32_to_f16_kernel(const float* __restrict__ src, _Float16* __restrict__ dst, int n8) {
  int i = blockIdx.x * blockDim.x + threadIdx.x;
  const int stride = gridDim.x * blockDim.x;
  for (; i < n8; i += stride) {
    const float4* s = (const float4*)src;
    float4 a = s[2 * i], b = s[2 * i + 1];
    v8h h;
    h[0] = (_Float16)a.x; h[1] = (_Float16)a.y; h[2] = (_Float16)a.z; h[3] = (_Float16)a.w;
    h[4] = (_Float16)b.x; h[5] = (_Float16)b.y; h[6] = (_Float16)b.z; h[7] = (_Float16)b.w;
    ((v8h*)dst)[i] = h;
  }
}

// ---------------- GEMM: C = A[M,1024] @ W[N,1024]^T ----------------
// MODE 0: f16 out, head-major  q[(b*H+h)*T + t][hd]      (Q, K)
// MODE 1: f16 out, transposed  vt[(b*H+h)*HD + hd][t]    (V)
// MODE 2: f32 out = (acc + bias[n]) * !pad[b,t]          (out-proj)
// block = 256 threads (8 waves): 4 waves along M x 2 along N -> 128M x 128N.
// B tile (128x32) is async-staged into double-buffered LDS, shared by all waves.
template <int MODE>
__global__ __launch_bounds__(256, 2)
void gemm_wmma_kernel(const _Float16* __restrict__ A,
                      const _Float16* __restrict__ W,
                      _Float16* __restrict__ outH,
                      float* __restrict__ outF,
                      const float* __restrict__ bias,
                      const unsigned char* __restrict__ pad) {
  __shared__ _Float16 Bs[2][128 * BLD];   // [local n][k], padded rows -> ~10KB x2

  const int K = D_;
  const int w = threadIdx.x >> 5;
  const int wm = w & 3, wn = w >> 2;
  const int m0 = blockIdx.x * 128 + wm * 32;
  const int n0 = blockIdx.y * 128 + wn * 64;
  const int nLoc0 = wn * 64;
  const int lane = threadIdx.x & 31, r = lane & 15, hf = lane >> 4;
  const int tid = threadIdx.x;

  // staging map: thread -> W row (blkN + tid/2), 16-half segment (tid&1)*16
  const int scol = tid >> 1;              // 0..127
  const int sseg = (tid & 1) * 16;        // 0 or 16 halfs
  const int blkN = blockIdx.y * 128;

  // prime buffer 0 with k0 = 0
#pragma unroll
  for (int q = 0; q < 2; ++q)
    async_ld_b128(lds_off_of(&Bs[0][scol * BLD + sseg + q * 8]),
                  W + (size_t)(blkN + scol) * K + sseg + q * 8);
  wait_async0();
  __syncthreads();

  v8f acc[2][4] = {};
  int buf = 0;
  for (int k0 = 0; k0 < K; k0 += 32) {
    if (k0 + 32 < K) {  // async-stage next B tile; prefetch next A lines
#pragma unroll
      for (int q = 0; q < 2; ++q)
        async_ld_b128(lds_off_of(&Bs[buf ^ 1][scol * BLD + sseg + q * 8]),
                      W + (size_t)(blkN + scol) * K + k0 + 32 + sseg + q * 8);
      __builtin_prefetch(A + (size_t)(m0 + r) * K + k0 + 32 + hf * 8, 0, 3);
    }
    v16h a0 = load_frag_a(A, K, m0, k0);
    v16h a1 = load_frag_a(A, K, m0 + 16, k0);
#pragma unroll
    for (int f = 0; f < 4; ++f) {
      v16h b = load_frag_b(&Bs[buf][0], BLD, nLoc0 + f * 16, 0);
      acc[0][f] = wmma32(a0, b, acc[0][f]);
      acc[1][f] = wmma32(a1, b, acc[1][f]);
    }
    wait_async0();
    __syncthreads();
    buf ^= 1;
  }

#pragma unroll
  for (int mi = 0; mi < 2; ++mi) {
    const int mrow0 = m0 + mi * 16;
    const int b = mrow0 >> 11;            // mrow0 / T_
    const int t0 = mrow0 & (T_ - 1);
    if (MODE == 0) {
#pragma unroll
      for (int f = 0; f < 4; ++f) {
        const int n = n0 + f * 16 + r;
        const int h = n >> 6, hd = n & 63;
        _Float16* qp = outH + ((size_t)(b * H_ + h) * T_) * HD_ + hd;
#pragma unroll
        for (int i = 0; i < 8; ++i) {
          const int t = t0 + hf * 8 + i;
          qp[(size_t)t * HD_] = (_Float16)acc[mi][f][i];
        }
      }
    } else if (MODE == 1) {
#pragma unroll
      for (int f = 0; f < 4; ++f) {
        const int n = n0 + f * 16 + r;
        const int h = n >> 6, hd = n & 63;
        v8h pk;
#pragma unroll
        for (int i = 0; i < 8; ++i) pk[i] = (_Float16)acc[mi][f][i];
        _Float16* vp = outH + ((size_t)(b * H_ + h) * HD_ + hd) * T_ + t0 + hf * 8;
        *(v8h*)vp = pk;   // 8 consecutive t values -> 16B store
      }
    } else {
#pragma unroll
      for (int f = 0; f < 4; ++f) {
        const int n = n0 + f * 16 + r;
        const float bn = bias[n];
#pragma unroll
        for (int i = 0; i < 8; ++i) {
          const int t = t0 + hf * 8 + i;
          float v = acc[mi][f][i] + bn;
          if (pad[b * T_ + t]) v = 0.f;
          outF[(size_t)(mrow0 + hf * 8 + i) * D_ + n] = v;
        }
      }
    }
  }
}

// ---------------- flash attention ----------------
// Block = 4 waves = 64 Q rows of one (b,h). K/V tiles (64x64) are async-copied
// into double-buffered LDS once per block and shared by all 4 waves.
__global__ __launch_bounds__(128, 2)
void flash_attn_kernel(const _Float16* __restrict__ Q,
                       const _Float16* __restrict__ Kd,
                       const _Float16* __restrict__ Vt,
                       const unsigned char* __restrict__ pad,
                       _Float16* __restrict__ attn) {
  __shared__ _Float16 KT[2][64 * KLD];    // [local j][hd], padded rows
  __shared__ _Float16 VT[2][64 * KLD];    // [hd][local j], padded rows
  __shared__ _Float16 Plds[4][16 * KLD];  // per-wave P staging

  const int w = threadIdx.x >> 5;
  const int lane = threadIdx.x & 31, r = lane & 15, hf = lane >> 4;
  const int tid = threadIdx.x;

  const int blocksPerBH = T_ / 64;         // 32
  const int bh = blockIdx.x / blocksPerBH;
  const int iBlk = (blockIdx.x % blocksPerBH) * 64;
  const int b = bh >> 4;                   // bh / H_
  const int h = bh & (H_ - 1);
  const int i0 = iBlk + w * 16;            // this wave's 16 rows
  const int myJend = i0 + 16;              // causal column bound (exclusive)
  const int jendBlk = iBlk + 64;           // block-wide bound

  const _Float16* Qb = Q  + (size_t)bh * T_ * HD_;
  const _Float16* Kb = Kd + (size_t)bh * T_ * HD_;
  const _Float16* Vb = Vt + (size_t)bh * HD_ * T_;

  // cooperative async staging: thread -> row tid/2, 32-half segment (tid&1)*32
  const int srow = tid >> 1;
  const int scol = (tid & 1) * 32;

  const v16h qa0 = load_frag_a(Qb, HD_, i0, 0);
  const v16h qa1 = load_frag_a(Qb, HD_, i0, 32);

  v8f o[4] = {};
  float mrow[8], lrow[8];
#pragma unroll
  for (int i = 0; i < 8; ++i) { mrow[i] = -__builtin_inff(); lrow[i] = 0.f; }

  const float scale = 0.125f;              // 1/sqrt(64)

  // prime buffer 0 with tile j0 = 0
#pragma unroll
  for (int q = 0; q < 4; ++q) {
    async_ld_b128(lds_off_of(&KT[0][srow * KLD + scol + q * 8]),
                  Kb + (size_t)srow * HD_ + scol + q * 8);
    async_ld_b128(lds_off_of(&VT[0][srow * KLD + scol + q * 8]),
                  Vb + (size_t)srow * T_ + scol + q * 8);
  }
  wait_async0();
  __syncthreads();

  int buf = 0;
  for (int j0 = 0; j0 < jendBlk; j0 += 64) {
    const int jn = j0 + 64;
    if (jn < jendBlk) {  // async-prefetch next K/V tile into the other buffer
#pragma unroll
      for (int q = 0; q < 4; ++q) {
        async_ld_b128(lds_off_of(&KT[buf ^ 1][srow * KLD + scol + q * 8]),
                      Kb + (size_t)(jn + srow) * HD_ + scol + q * 8);
        async_ld_b128(lds_off_of(&VT[buf ^ 1][srow * KLD + scol + q * 8]),
                      Vb + (size_t)srow * T_ + jn + scol + q * 8);
      }
    }

    if (j0 < myJend) {   // skip tiles entirely above this wave's diagonal
      const _Float16* kt = &KT[buf][0];
      const _Float16* vt = &VT[buf][0];

      // ---- S = Q @ K^T  (16 x 64 tile, four 16x16 C fragments, from LDS) ----
      v8f s[4];
#pragma unroll
      for (int f = 0; f < 4; ++f) {
        v8f t = {};
        v16h kb = load_frag_b(kt, KLD, f * 16, 0);   t = wmma32(qa0, kb, t);
        kb      = load_frag_b(kt, KLD, f * 16, 32);  t = wmma32(qa1, kb, t);
        s[f] = t;
      }

      int  c[4];
      bool pd[4];
#pragma unroll
      for (int f = 0; f < 4; ++f) {
        c[f] = j0 + f * 16 + r;
        pd[f] = pad[b * T_ + c[f]] != 0;
      }

      // ---- scale + causal/pad mask + online softmax ----
#pragma unroll
      for (int i = 0; i < 8; ++i) {
        const int row = i0 + hf * 8 + i;
        float mx = -__builtin_inff();
#pragma unroll
        for (int f = 0; f < 4; ++f) {
          float v = s[f][i] * scale;
          if (c[f] > row || pd[f]) v = -__builtin_inff();
          s[f][i] = v;
          mx = fmaxf(mx, v);
        }
        mx = fmaxf(mx, __shfl_xor(mx, 1, 16));
        mx = fmaxf(mx, __shfl_xor(mx, 2, 16));
        mx = fmaxf(mx, __shfl_xor(mx, 4, 16));
        mx = fmaxf(mx, __shfl_xor(mx, 8, 16));
        const float mnew = fmaxf(mrow[i], mx);
        const float corr = __expf(mrow[i] - mnew);
        float rs = 0.f;
#pragma unroll
        for (int f = 0; f < 4; ++f) {
          const float e = __expf(s[f][i] - mnew);
          s[f][i] = e;
          rs += e;
        }
        rs += __shfl_xor(rs, 1, 16);
        rs += __shfl_xor(rs, 2, 16);
        rs += __shfl_xor(rs, 4, 16);
        rs += __shfl_xor(rs, 8, 16);
        lrow[i] = lrow[i] * corr + rs;
        mrow[i] = mnew;
#pragma unroll
        for (int f = 0; f < 4; ++f) o[f][i] *= corr;
      }

      // ---- transpose P (C-layout) -> A-fragments via per-wave LDS ----
      _Float16* pl = Plds[w];
#pragma unroll
      for (int i = 0; i < 8; ++i) {
        const int rowl = hf * 8 + i;
#pragma unroll
        for (int f = 0; f < 4; ++f)
          pl[rowl * KLD + f * 16 + r] = (_Float16)s[f][i];
      }
      __asm volatile("" ::: "memory");   // keep LDS stores before reloads
      const v16h pa0 = load_frag_a(pl, KLD, 0, 0);
      const v16h pa1 = load_frag_a(pl, KLD, 0, 32);
      __asm volatile("" ::: "memory");

      // ---- O += P @ V  (V tile from LDS, columns contiguous) ----
#pragma unroll
      for (int f = 0; f < 4; ++f) {
        v16h vb = load_frag_b(vt, KLD, f * 16, 0);    o[f] = wmma32(pa0, vb, o[f]);
        vb      = load_frag_b(vt, KLD, f * 16, 32);   o[f] = wmma32(pa1, vb, o[f]);
      }
    }

    wait_async0();     // our async loads for the next buffer have landed
    __syncthreads();   // everyone done reading current buffer
    buf ^= 1;
  }

  // ---- normalize and store 16x64 output tile as f16 ----
#pragma unroll
  for (int i = 0; i < 8; ++i) {
    const float inv = 1.f / lrow[i];
#pragma unroll
    for (int f = 0; f < 4; ++f) o[f][i] *= inv;
  }
#pragma unroll
  for (int f = 0; f < 4; ++f) {
    const int hd = f * 16 + r;
#pragma unroll
    for (int i = 0; i < 8; ++i) {
      const int t = i0 + hf * 8 + i;
      attn[((size_t)(b * T_ + t)) * D_ + h * HD_ + hd] = (_Float16)o[f][i];
    }
  }
}

// ---------------- launch ----------------
extern "C" void kernel_launch(void* const* d_in, const int* in_sizes, int n_in,
                              void* d_out, int out_size, void* d_ws, size_t ws_size,
                              hipStream_t stream) {
  const float* x  = (const float*)d_in[0];
  const unsigned char* pad = (const unsigned char*)d_in[1];
  const float* Wq = (const float*)d_in[2];
  const float* Wk = (const float*)d_in[3];
  const float* Wv = (const float*)d_in[4];
  const float* Wo = (const float*)d_in[5];
  const float* bo = (const float*)d_in[6];
  float* out = (float*)d_out;

  char* ws = (char*)d_ws;
  size_t off = 0;
  const size_t xBytes = (size_t)B_ * T_ * D_ * sizeof(_Float16);   // 16 MB
  const size_t wBytes = (size_t)D_ * D_ * sizeof(_Float16);        // 2 MB
  _Float16* xh  = (_Float16*)(ws + off); off += xBytes;
  _Float16* wqh = (_Float16*)(ws + off); off += wBytes;
  _Float16* wkh = (_Float16*)(ws + off); off += wBytes;
  _Float16* wvh = (_Float16*)(ws + off); off += wBytes;
  _Float16* woh = (_Float16*)(ws + off); off += wBytes;
  _Float16* qh  = (_Float16*)(ws + off); off += xBytes;            // [B,H,T,HD]
  _Float16* kh  = (_Float16*)(ws + off); off += xBytes;            // [B,H,T,HD]
  _Float16* vt  = (_Float16*)(ws + off); off += xBytes;            // [B,H,HD,T]
  _Float16* ah  = (_Float16*)(ws + off); off += xBytes;            // [B,T,D]

  // 1) convert inputs to f16
  f32_to_f16_kernel<<<2048, 256, 0, stream>>>(x,  xh,  (B_ * T_ * D_) / 8);
  f32_to_f16_kernel<<<512,  256, 0, stream>>>(Wq, wqh, (D_ * D_) / 8);
  f32_to_f16_kernel<<<512,  256, 0, stream>>>(Wk, wkh, (D_ * D_) / 8);
  f32_to_f16_kernel<<<512,  256, 0, stream>>>(Wv, wvh, (D_ * D_) / 8);
  f32_to_f16_kernel<<<512,  256, 0, stream>>>(Wo, woh, (D_ * D_) / 8);

  // 2) projections (M=8192, N=1024, K=1024), 128x128 block tiles, staged B
  dim3 g((B_ * T_) / 128, D_ / 128), blk(256);
  gemm_wmma_kernel<0><<<g, blk, 0, stream>>>(xh, wqh, qh, nullptr, nullptr, nullptr);
  gemm_wmma_kernel<0><<<g, blk, 0, stream>>>(xh, wkh, kh, nullptr, nullptr, nullptr);
  gemm_wmma_kernel<1><<<g, blk, 0, stream>>>(xh, wvh, vt, nullptr, nullptr, nullptr);

  // 3) causal flash attention: 64-row blocks, shared async-staged K/V tiles
  flash_attn_kernel<<<B_ * H_ * (T_ / 64), 128, 0, stream>>>(qh, kh, vt, pad, ah);

  // 4) output projection + bias + padding zero
  gemm_wmma_kernel<2><<<g, blk, 0, stream>>>(ah, woh, nullptr, out, bo, pad);
}
